// Polarizable_Interaction_MP_Block_62122406969666
// MI455X (gfx1250) — compile-verified, matrix operand order checked by reading, not run on hardware
//
#include <hip/hip_runtime.h>

// ---------------- types ----------------
typedef __attribute__((ext_vector_type(16))) __bf16 v16bf;
typedef __attribute__((ext_vector_type(8)))  float  v8f;

#define N_NODES 50000
#define N_EDGES 800000
#define TILES   (N_EDGES / 16)   // 50000
#define WPB     4                // waves per block
#define BLK     (WPB * 32)

union FragU { v16bf v; unsigned short s[16]; uint4 q[2]; };

__device__ inline unsigned short f2bf(float f) {
  unsigned u = __float_as_uint(f);
  unsigned r = u + 0x7FFFu + ((u >> 16) & 1u);   // round-to-nearest-even
  return (unsigned short)(r >> 16);
}

// pack 4 floats -> 4 bf16, single 8B LDS store
__device__ inline void store_bf16x4(unsigned short* p, float4 f) {
  ushort4 u;
  u.x = f2bf(f.x); u.y = f2bf(f.y); u.z = f2bf(f.z); u.w = f2bf(f.w);
  *(ushort4*)p = u;
}

__device__ inline float silu_f(float x) {
  // fast: v_exp_f32 + v_rcp_f32 instead of IEEE division chain
  return x * __builtin_amdgcn_rcpf(1.f + __expf(-x));
}

__device__ inline v8f wmma_bf16(v16bf a, v16bf b, v8f c) {
  // D = A(16x32 bf16) * B(32x16 bf16) + C(16x16 f32)
  return __builtin_amdgcn_wmma_f32_16x16x32_bf16(false, a, false, b, (short)0, c,
                                                 false, false);
}

// A fragment from a row-major bf16 LDS tile (16 x ld), k-tile kt.
// ISA 16-bit A layout: lane m=L&15, hi=L>>4; K(i,hi) = (i&8)*2 + 8*hi + (i&7)
// => two contiguous 8-element (16B) chunks per lane.
__device__ inline v16bf frag_a(const unsigned short* lds, int ld, int kt) {
  const int lane = threadIdx.x & 31;
  const int m = lane & 15, hi = lane >> 4;
  FragU u;
  const unsigned short* p = lds + m * ld + kt * 32 + hi * 8;
  u.q[0] = *(const uint4*)(p);
  u.q[1] = *(const uint4*)(p + 16);
  return u.v;
}

// B fragment from pre-packed global weights: 32 lanes x 16 bf16, lane-major.
__device__ inline v16bf frag_b(const unsigned short* g) {
  const int lane = threadIdx.x & 31;
  FragU u;
  const uint4* p = (const uint4*)(g + (size_t)lane * 16);
  u.q[0] = p[0];
  u.q[1] = p[1];
  return u.v;
}

__device__ inline void atomicMaxF(float* a, float v) {
  if (v >= 0.f) atomicMax((int*)a, __float_as_int(v));
  else          atomicMin((unsigned int*)a, __float_as_uint(v));
}

// ---------------- weight packing ----------------
// Pack f32 weight [Ksrc x N] (row-major, per-head) into bf16 B-fragments.
// Fragment (kt,nt): 512 bf16 = lane*16 + i, value = W[kt*32 + K(i,hi)][nt*16 + nn]
__global__ void k_pack(const float* __restrict__ src, unsigned short* __restrict__ dst,
                       int Ksrc, int N, int ktCnt, int ntCnt) {
  const int frag = blockIdx.x, head = blockIdx.y;
  const int kt = frag / ntCnt, nt = frag % ntCnt;
  const int lane = threadIdx.x;
  const int nn = lane & 15, hi = lane >> 4;
  const float* s = src + (size_t)head * Ksrc * N;
  unsigned short* d = dst + ((size_t)(head * ktCnt * ntCnt + frag)) * 512 + (size_t)lane * 16;
  const int n = nt * 16 + nn;
#pragma unroll
  for (int i = 0; i < 16; ++i) {
    int k = kt * 32 + ((i & 8) << 1) + hi * 8 + (i & 7);
    float v = (k < Ksrc && n < N) ? s[(size_t)k * N + n] : 0.f;
    d[i] = f2bf(v);
  }
}

// ---------------- stage 1: node MLP + q/k + attention logits ----------------
__global__ __launch_bounds__(BLK) void k_edge_att(
    const float* __restrict__ node_feat, const float* __restrict__ edge_attr,
    const int* __restrict__ edges,
    const float* __restrict__ nc_b1, const float* __restrict__ nc_b2,
    const unsigned short* __restrict__ p_w1, const unsigned short* __restrict__ p_w2,
    const unsigned short* __restrict__ p_wq, const unsigned short* __restrict__ p_wk,
    unsigned short* __restrict__ node_out, float* __restrict__ att,
    float* __restrict__ nmax)
{
  __shared__ __align__(16) unsigned short sH[WPB][16 * 96];
  __shared__ __align__(16) unsigned short sM[WPB][16 * 96];
  const int wave = threadIdx.x >> 5, lane = threadIdx.x & 31;
  const int nn = lane & 15, hi = lane >> 4;
  const int tile = blockIdx.x * WPB + wave;
  unsigned short* bufH = sH[wave];
  unsigned short* bufM = sM[wave];
  const int e0 = tile * 16;
  const int* col = edges + N_EDGES;

  // stage H = [node_feat[col] (64) | edge_attr (16) | zero pad (16)] as bf16
  // node_feat gather: 16 rows x 16 float4 (global_load_b128 + ds_store_b64)
  for (int idx = lane; idx < 256; idx += 32) {
    int r = idx >> 4, c4 = idx & 15;
    float4 f = *(const float4*)(node_feat + (size_t)col[e0 + r] * 64 + c4 * 4);
    store_bf16x4(bufH + r * 96 + c4 * 4, f);
  }
  // edge_attr: 16 rows x 4 float4
  for (int idx = lane; idx < 64; idx += 32) {
    int r = idx >> 2, c4 = idx & 3;
    float4 f = *(const float4*)(edge_attr + (size_t)(e0 + r) * 16 + c4 * 4);
    store_bf16x4(bufH + r * 96 + 64 + c4 * 4, f);
  }
  // zero pad cols 80..95
  for (int idx = lane; idx < 64; idx += 32) {
    int r = idx >> 2, c4 = idx & 3;
    ushort4 z4; z4.x = z4.y = z4.z = z4.w = 0;
    *(ushort4*)(bufH + r * 96 + 80 + c4 * 4) = z4;
  }
  __syncthreads();

  // h1 = silu(H @ nc_w1 + b1)  (K=96 padded, N=32) -> bufM
  {
    v16bf a0 = frag_a(bufH, 96, 0), a1 = frag_a(bufH, 96, 1), a2 = frag_a(bufH, 96, 2);
#pragma unroll
    for (int nt = 0; nt < 2; ++nt) {
      v8f c = {0.f,0.f,0.f,0.f,0.f,0.f,0.f,0.f};
      c = wmma_bf16(a0, frag_b(p_w1 + (size_t)(0 * 2 + nt) * 512), c);
      c = wmma_bf16(a1, frag_b(p_w1 + (size_t)(1 * 2 + nt) * 512), c);
      c = wmma_bf16(a2, frag_b(p_w1 + (size_t)(2 * 2 + nt) * 512), c);
      float b = nc_b1[nt * 16 + nn];
#pragma unroll
      for (int v = 0; v < 8; ++v)
        bufM[(v + 8 * hi) * 96 + nt * 16 + nn] = f2bf(silu_f(c[v] + b));
    }
  }
  __syncthreads();

  // node_out = h1 @ nc_w2 + b2  (K=32, N=96) -> bufH (reused)
  {
    v16bf a = frag_a(bufM, 96, 0);
#pragma unroll
    for (int nt = 0; nt < 6; ++nt) {
      v8f c = {0.f,0.f,0.f,0.f,0.f,0.f,0.f,0.f};
      c = wmma_bf16(a, frag_b(p_w2 + (size_t)nt * 512), c);
      float b = nc_b2[nt * 16 + nn];
#pragma unroll
      for (int v = 0; v < 8; ++v)
        bufH[(v + 8 * hi) * 96 + nt * 16 + nn] = f2bf(c[v] + b);
    }
  }
  __syncthreads();

  // spill node_out tile (bf16) to workspace, b128 copies
  {
    const uint4* s = (const uint4*)bufH;
    uint4* d = (uint4*)(node_out + (size_t)e0 * 96);
    for (int i = lane; i < 192; i += 32) d[i] = s[i];
  }

  // per-head q,k via WMMA; att logit = (q . k) / sqrt(96)
  {
    v16bf a0 = frag_a(bufH, 96, 0), a1 = frag_a(bufH, 96, 1), a2 = frag_a(bufH, 96, 2);
    const float rs = 0.102062072615966f; // 1/sqrt(96)
    for (int h = 0; h < 4; ++h) {
      float acc[8] = {0.f,0.f,0.f,0.f,0.f,0.f,0.f,0.f};
      for (int nt = 0; nt < 6; ++nt) {
        v8f qc = {0.f,0.f,0.f,0.f,0.f,0.f,0.f,0.f};
        v8f kc = {0.f,0.f,0.f,0.f,0.f,0.f,0.f,0.f};
        const size_t base = (size_t)h * 18;
        qc = wmma_bf16(a0, frag_b(p_wq + (base + 0 * 6 + nt) * 512), qc);
        qc = wmma_bf16(a1, frag_b(p_wq + (base + 1 * 6 + nt) * 512), qc);
        qc = wmma_bf16(a2, frag_b(p_wq + (base + 2 * 6 + nt) * 512), qc);
        kc = wmma_bf16(a0, frag_b(p_wk + (base + 0 * 6 + nt) * 512), kc);
        kc = wmma_bf16(a1, frag_b(p_wk + (base + 1 * 6 + nt) * 512), kc);
        kc = wmma_bf16(a2, frag_b(p_wk + (base + 2 * 6 + nt) * 512), kc);
#pragma unroll
        for (int v = 0; v < 8; ++v) acc[v] += qc[v] * kc[v];
      }
      // reduce over the 16 lanes holding the same rows
#pragma unroll
      for (int off = 1; off < 16; off <<= 1) {
#pragma unroll
        for (int v = 0; v < 8; ++v) acc[v] += __shfl_xor(acc[v], off, 32);
      }
      if (nn == 0) {
#pragma unroll
        for (int v = 0; v < 8; ++v) {
          int m = v + 8 * hi;
          int e = e0 + m;
          float logit = acc[v] * rs;
          att[(size_t)e * 4 + h] = logit;
          atomicMaxF(&nmax[(size_t)edges[e] * 4 + h], logit);
        }
      }
    }
  }
}

// ---------------- stage 2: exp + segment sums + counts ----------------
__global__ void k_softmax(const int* __restrict__ edges, float* __restrict__ att,
                          const float* __restrict__ nmax, float* __restrict__ nsum,
                          float* __restrict__ cnt)
{
  int idx = blockIdx.x * blockDim.x + threadIdx.x;
  if (idx >= N_EDGES * 4) return;
  int e = idx >> 2, h = idx & 3;
  int r = edges[e];
  float m = nmax[(size_t)r * 4 + h];
  if (!(fabsf(m) < 1e30f)) m = 0.f;           // NaN / inf guard (matches ref isfinite)
  float ex = __expf(att[idx] - m);
  att[idx] = ex;
  atomicAdd(&nsum[(size_t)r * 4 + h], ex);
  if (h == 0) atomicAdd(&cnt[r], 1.f);
}

// ---------------- stage 3: v, z, out MLP, rbf, mix, scatter ----------------
__global__ __launch_bounds__(BLK) void k_edge_out(
    const int* __restrict__ edges, const float* __restrict__ vec_feat,
    const float* __restrict__ coord_diff, const float* __restrict__ rbf,
    const unsigned short* __restrict__ node_out,
    const float* __restrict__ att, const float* __restrict__ nsum,
    const unsigned short* __restrict__ p_wv, const unsigned short* __restrict__ p_ow1,
    const unsigned short* __restrict__ p_ow2, const unsigned short* __restrict__ p_rbfw,
    const float* __restrict__ out_b1, const float* __restrict__ out_b2,
    const float* __restrict__ rbf_b,
    float* __restrict__ acc_node, float* __restrict__ acc_vec)
{
  __shared__ __align__(16) unsigned short sA[WPB][16 * 96];
  __shared__ __align__(16) unsigned short sB[WPB][16 * 96];
  __shared__ float sAtt[WPB][16 * 4];
  __shared__ int   sRow[WPB][16];
  __shared__ int   sCol[WPB][16];
  const int wave = threadIdx.x >> 5, lane = threadIdx.x & 31;
  const int nn = lane & 15, hi = lane >> 4;
  const int tile = blockIdx.x * WPB + wave;
  unsigned short* bufA = sA[wave];
  unsigned short* bufB = sB[wave];
  const int e0 = tile * 16;

  // prefetch next tile's node_out block (contiguous in workspace)
  __builtin_prefetch(node_out + (size_t)(e0 + 16) * 96, 0, 3);

  // reload node_out tile to LDS
  {
    const uint4* s = (const uint4*)(node_out + (size_t)e0 * 96);
    uint4* d = (uint4*)bufA;
    for (int i = lane; i < 192; i += 32) d[i] = s[i];
  }
  if (lane < 16) {
    int e = e0 + lane;
    int r = edges[e];
    sRow[wave][lane] = r;
    sCol[wave][lane] = edges[N_EDGES + e];
#pragma unroll
    for (int h = 0; h < 4; ++h)
      sAtt[wave][lane * 4 + h] = att[(size_t)e * 4 + h] / nsum[(size_t)r * 4 + h];
  }
  __syncthreads();

  // z = sum_h att_h * (node_out @ wv_h)  -> bufB (bf16)
  {
    v16bf a0 = frag_a(bufA, 96, 0), a1 = frag_a(bufA, 96, 1), a2 = frag_a(bufA, 96, 2);
    for (int nt = 0; nt < 6; ++nt) {
      float zc[8] = {0.f,0.f,0.f,0.f,0.f,0.f,0.f,0.f};
      for (int h = 0; h < 4; ++h) {
        v8f vc = {0.f,0.f,0.f,0.f,0.f,0.f,0.f,0.f};
        const size_t base = (size_t)h * 18;
        vc = wmma_bf16(a0, frag_b(p_wv + (base + 0 * 6 + nt) * 512), vc);
        vc = wmma_bf16(a1, frag_b(p_wv + (base + 1 * 6 + nt) * 512), vc);
        vc = wmma_bf16(a2, frag_b(p_wv + (base + 2 * 6 + nt) * 512), vc);
#pragma unroll
        for (int v = 0; v < 8; ++v)
          zc[v] += sAtt[wave][(v + 8 * hi) * 4 + h] * vc[v];
      }
#pragma unroll
      for (int v = 0; v < 8; ++v)
        bufB[(v + 8 * hi) * 96 + nt * 16 + nn] = f2bf(zc[v]);
    }
  }
  __syncthreads();

  // t = silu(z @ out_w1 + b1) -> bufA (reused)
  {
    v16bf a0 = frag_a(bufB, 96, 0), a1 = frag_a(bufB, 96, 1), a2 = frag_a(bufB, 96, 2);
    for (int nt = 0; nt < 6; ++nt) {
      v8f c = {0.f,0.f,0.f,0.f,0.f,0.f,0.f,0.f};
      c = wmma_bf16(a0, frag_b(p_ow1 + (size_t)(0 * 6 + nt) * 512), c);
      c = wmma_bf16(a1, frag_b(p_ow1 + (size_t)(1 * 6 + nt) * 512), c);
      c = wmma_bf16(a2, frag_b(p_ow1 + (size_t)(2 * 6 + nt) * 512), c);
      float b = out_b1[nt * 16 + nn];
#pragma unroll
      for (int v = 0; v < 8; ++v)
        bufA[(v + 8 * hi) * 96 + nt * 16 + nn] = f2bf(silu_f(c[v] + b));
    }
  }
  __syncthreads();

  // stage rbf tile (16 x 32, padded from 20) in bufB: 16 rows x 5 float4
  for (int idx = lane; idx < 80; idx += 32) {
    int r = idx / 5, c4 = idx % 5;
    float4 f = *(const float4*)(rbf + (size_t)(e0 + r) * 20 + c4 * 4);
    store_bf16x4(bufB + r * 32 + c4 * 4, f);
  }
  for (int idx = lane; idx < 48; idx += 32) {
    int r = idx / 3, c4 = idx % 3;
    ushort4 z4; z4.x = z4.y = z4.z = z4.w = 0;
    *(ushort4*)(bufB + r * 32 + 20 + c4 * 4) = z4;
  }
  __syncthreads();

  // o = t @ out_w2 + b2 ; rb = rbf @ rbf_w + rbf_b ; mix = o * rb ; scatter
  {
    v16bf t0 = frag_a(bufA, 96, 0), t1 = frag_a(bufA, 96, 1), t2 = frag_a(bufA, 96, 2);
    v16bf ar = frag_a(bufB, 32, 0);
    for (int nt = 0; nt < 6; ++nt) {
      v8f oc = {0.f,0.f,0.f,0.f,0.f,0.f,0.f,0.f};
      v8f rc = {0.f,0.f,0.f,0.f,0.f,0.f,0.f,0.f};
      oc = wmma_bf16(t0, frag_b(p_ow2 + (size_t)(0 * 6 + nt) * 512), oc);
      oc = wmma_bf16(t1, frag_b(p_ow2 + (size_t)(1 * 6 + nt) * 512), oc);
      oc = wmma_bf16(t2, frag_b(p_ow2 + (size_t)(2 * 6 + nt) * 512), oc);
      rc = wmma_bf16(ar, frag_b(p_rbfw + (size_t)nt * 512), rc);
      const int n = nt * 16 + nn;
      const float b2v = out_b2[n], rbv = rbf_b[n];
#pragma unroll
      for (int v = 0; v < 8; ++v) {
        int m = v + 8 * hi;
        int e = e0 + m;
        float mix = (oc[v] + b2v) * (rc[v] + rbv);
        int r = sRow[wave][m];
        if (n < 32) {                 // mes_vec += vec_feat[col] * mix[:, :32]
          int cidx = sCol[wave][m];
          const float* vf = vec_feat + (size_t)cidx * 96 + n * 3;
          atomicAdd(&acc_vec[(size_t)r * 96 + n * 3 + 0], vf[0] * mix);
          atomicAdd(&acc_vec[(size_t)r * 96 + n * 3 + 1], vf[1] * mix);
          atomicAdd(&acc_vec[(size_t)r * 96 + n * 3 + 2], vf[2] * mix);
        } else if (n < 64) {          // mes_node = mix[:, 32:64]
          atomicAdd(&acc_node[(size_t)r * 32 + (n - 32)], mix);
        } else {                      // mes_vec += coord_diff * mix[:, 64:96]
          const float* cd = coord_diff + (size_t)e * 3;
          int j = n - 64;
          atomicAdd(&acc_vec[(size_t)r * 96 + j * 3 + 0], cd[0] * mix);
          atomicAdd(&acc_vec[(size_t)r * 96 + j * 3 + 1], cd[1] * mix);
          atomicAdd(&acc_vec[(size_t)r * 96 + j * 3 + 2], cd[2] * mix);
        }
      }
    }
  }
}

// ---------------- stage 4: segment mean + output ----------------
__global__ void k_finalize(const float* __restrict__ acc_node,
                           const float* __restrict__ acc_vec,
                           const float* __restrict__ cnt, float* __restrict__ out)
{
  int idx = blockIdx.x * blockDim.x + threadIdx.x;
  const int NN32 = N_NODES * 32;
  const int TOT = N_NODES * 128;
  if (idx >= TOT) return;
  if (idx < NN32) {
    float c = fmaxf(cnt[idx / 32], 1.f);
    out[idx] = acc_node[idx] / c;
  } else {
    int j = idx - NN32;
    float c = fmaxf(cnt[j / 96], 1.f);
    out[idx] = acc_vec[j] / c;
  }
}

// ---------------- host ----------------
extern "C" void kernel_launch(void* const* d_in, const int* in_sizes, int n_in,
                              void* d_out, int out_size, void* d_ws, size_t ws_size,
                              hipStream_t stream) {
  (void)in_sizes; (void)n_in; (void)out_size; (void)ws_size;
  const float* node_feat  = (const float*)d_in[0];
  const float* vec_feat   = (const float*)d_in[1];
  const int*   edges      = (const int*)d_in[2];
  const float* edge_attr  = (const float*)d_in[3];
  const float* coord_diff = (const float*)d_in[4];
  const float* rbf        = (const float*)d_in[5];
  const float* nc_w1 = (const float*)d_in[6];
  const float* nc_b1 = (const float*)d_in[7];
  const float* nc_w2 = (const float*)d_in[8];
  const float* nc_b2 = (const float*)d_in[9];
  const float* rbf_w = (const float*)d_in[10];
  const float* rbf_b = (const float*)d_in[11];
  const float* wq    = (const float*)d_in[12];
  const float* wk    = (const float*)d_in[13];
  const float* wv    = (const float*)d_in[14];
  const float* out_w1 = (const float*)d_in[15];
  const float* out_b1 = (const float*)d_in[16];
  const float* out_w2 = (const float*)d_in[17];
  const float* out_b2 = (const float*)d_in[18];

  char* ws = (char*)d_ws;
  size_t off = 0;
  auto take = [&](size_t bytes) {
    char* p = ws + off;
    off = (off + bytes + 255) & ~(size_t)255;
    return p;
  };
  unsigned short* p_w1  = (unsigned short*)take(6 * 1024);
  unsigned short* p_w2  = (unsigned short*)take(6 * 1024);
  unsigned short* p_rbf = (unsigned short*)take(6 * 1024);
  unsigned short* p_wq  = (unsigned short*)take(72 * 1024);
  unsigned short* p_wk  = (unsigned short*)take(72 * 1024);
  unsigned short* p_wv  = (unsigned short*)take(72 * 1024);
  unsigned short* p_ow1 = (unsigned short*)take(18 * 1024);
  unsigned short* p_ow2 = (unsigned short*)take(18 * 1024);
  unsigned short* w_nodeout = (unsigned short*)take((size_t)N_EDGES * 96 * 2);
  float* w_att  = (float*)take((size_t)N_EDGES * 4 * 4);
  float* w_nmax = (float*)take((size_t)N_NODES * 4 * 4);
  float* w_nsum = (float*)take((size_t)N_NODES * 4 * 4);
  float* w_accn = (float*)take((size_t)N_NODES * 32 * 4);
  float* w_accv = (float*)take((size_t)N_NODES * 96 * 4);
  float* w_cnt  = (float*)take((size_t)N_NODES * 4);

  hipMemsetAsync(w_nmax, 0xFF, (size_t)N_NODES * 16, stream); // NaN sentinel
  hipMemsetAsync(w_nsum, 0, (size_t)N_NODES * 16, stream);
  hipMemsetAsync(w_accn, 0, (size_t)N_NODES * 128, stream);
  hipMemsetAsync(w_accv, 0, (size_t)N_NODES * 384, stream);
  hipMemsetAsync(w_cnt,  0, (size_t)N_NODES * 4, stream);

  k_pack<<<dim3(6, 1),  32, 0, stream>>>(nc_w1,  p_w1,  80, 32, 3, 2);
  k_pack<<<dim3(6, 1),  32, 0, stream>>>(nc_w2,  p_w2,  32, 96, 1, 6);
  k_pack<<<dim3(6, 1),  32, 0, stream>>>(rbf_w,  p_rbf, 20, 96, 1, 6);
  k_pack<<<dim3(18, 4), 32, 0, stream>>>(wq,     p_wq,  96, 96, 3, 6);
  k_pack<<<dim3(18, 4), 32, 0, stream>>>(wk,     p_wk,  96, 96, 3, 6);
  k_pack<<<dim3(18, 4), 32, 0, stream>>>(wv,     p_wv,  96, 96, 3, 6);
  k_pack<<<dim3(18, 1), 32, 0, stream>>>(out_w1, p_ow1, 96, 96, 3, 6);
  k_pack<<<dim3(18, 1), 32, 0, stream>>>(out_w2, p_ow2, 96, 96, 3, 6);

  k_edge_att<<<TILES / WPB, BLK, 0, stream>>>(
      node_feat, edge_attr, edges, nc_b1, nc_b2,
      p_w1, p_w2, p_wq, p_wk, w_nodeout, w_att, w_nmax);

  k_softmax<<<(N_EDGES * 4 + 255) / 256, 256, 0, stream>>>(
      edges, w_att, w_nmax, w_nsum, w_cnt);

  k_edge_out<<<TILES / WPB, BLK, 0, stream>>>(
      edges, vec_feat, coord_diff, rbf, w_nodeout, w_att, w_nsum,
      p_wv, p_ow1, p_ow2, p_rbf, out_b1, out_b2, rbf_b, w_accn, w_accv);

  k_finalize<<<(N_NODES * 128 + 255) / 256, 256, 0, stream>>>(
      w_accn, w_accv, w_cnt, (float*)d_out);
}
